// FusionLayer_74998718923010
// MI455X (gfx1250) — matrix operand order repaired and checked by reading, not compiled, not run on hardware
//
#include <hip/hip_runtime.h>

// ---------------------------------------------------------------------------
// FusionLayer for MI455X (gfx1250): latent attention + Chebyshev graph conv
// B=1024, N=256, C=128, L=4, K=3.  bf16 WMMA (16x16x32) with f32 accumulate.
// 4-wide accumulator blocking (32 VGPRs) to stay spill-free under 256 VGPRs.
// ---------------------------------------------------------------------------

#define BATCH 1024
#define NN    256
#define CC    128
#define LDX   136   // row stride (elements) for row-major 256x128 bf16 tiles (272B, 16B aligned)
#define LDT   264   // row stride (elements) for transposed 128x256 bf16 tiles (528B, 16B aligned)

// ws layout (in unsigned short elements):
#define WS_ADJ  0                       // 256*256 bf16 adjacency
#define WS_WT   65536                   // 3*128*128 bf16: (W0-W2)^T, W1^T, (2W2)^T for time
#define WS_WF   (65536 + 49152)         // same for fre
#define WS_BYTES 327680

typedef __attribute__((ext_vector_type(16))) __bf16 v16bf;
typedef __attribute__((ext_vector_type(8)))  float  v8f;

union FragBF { v16bf v; uint4 q[2]; };
union Pack8  { unsigned short h[8]; uint4 q; };
union Pack4  { unsigned short h[4]; uint2 q; };

static __device__ __forceinline__ unsigned short f2bf(float f) {
    union { float f; unsigned u; } v; v.f = f;
    unsigned r = v.u + 0x7FFFu + ((v.u >> 16) & 1u);   // round-to-nearest-even
    return (unsigned short)(r >> 16);
}
static __device__ __forceinline__ float bf2f(unsigned short h) {
    union { unsigned u; float f; } v; v.u = ((unsigned)h) << 16;
    return v.f;
}
static __device__ __forceinline__ float wred32(float v) {
    #pragma unroll
    for (int m = 16; m > 0; m >>= 1) v += __shfl_xor(v, m, 32);
    return v;
}

// ---------------------------------------------------------------------------
// Prep: adj -> bf16; fold Chebyshev weights into transposed bf16 matrices.
// ---------------------------------------------------------------------------
__global__ void fl_prep(const float* __restrict__ adj,
                        const float* __restrict__ Wt,
                        const float* __restrict__ Wf,
                        unsigned short* __restrict__ ws) {
    int t = blockIdx.x * blockDim.x + threadIdx.x;        // 65536 threads
    ws[WS_ADJ + t] = f2bf(adj[t]);
    if (t < 16384) {
        int c = t >> 7, d = t & 127;
        int src = c * 128 + d;        // W[k][c][d]
        int dst = d * 128 + c;        // transposed
        {
            float w0 = Wt[src], w1 = Wt[16384 + src], w2 = Wt[32768 + src];
            ws[WS_WT +          dst] = f2bf(w0 - w2);
            ws[WS_WT + 16384 +  dst] = f2bf(w1);
            ws[WS_WT + 32768 +  dst] = f2bf(2.0f * w2);
        }
        {
            float w0 = Wf[src], w1 = Wf[16384 + src], w2 = Wf[32768 + src];
            ws[WS_WF +          dst] = f2bf(w0 - w2);
            ws[WS_WF + 16384 +  dst] = f2bf(w1);
            ws[WS_WF + 32768 +  dst] = f2bf(2.0f * w2);
        }
    }
}

// ---------------------------------------------------------------------------
// Main fused kernel: one workgroup per batch, 8 wave32s.
// ---------------------------------------------------------------------------
__global__ __launch_bounds__(256, 1)
void fl_main(const float* __restrict__ gtime,
             const float* __restrict__ gfre,
             const float* __restrict__ glat,
             const float* __restrict__ gscale_t,
             const float* __restrict__ gscale_f,
             const float* __restrict__ gbt,
             const float* __restrict__ gbf,
             const unsigned short* __restrict__ wsB,
             float* __restrict__ gout) {

    __shared__ unsigned short sX  [NN * LDX];   // x (row-major bf16), updated in place
    __shared__ unsigned short sR2 [NN * LDX];   // x^T (stride LDT) then u (stride LDX)
    __shared__ unsigned short sT1 [NN * LDX];   // initially fre, later t1 (row-major)
    __shared__ unsigned short sT1T[CC * LDT];   // t1^T
    __shared__ float sLat  [4 * CC];
    __shared__ float sFused[4 * CC];
    __shared__ float sLog  [4 * 2 * NN];

    const int b    = blockIdx.x;
    const int tid  = threadIdx.x;
    const int lane = tid & 31;
    const int wave = tid >> 5;

    const float* timeB = gtime + (size_t)b * NN * CC;
    const float* freB  = gfre  + (size_t)b * NN * CC;
    const unsigned short* adjB = wsB + WS_ADJ;

    // warm L2 with the bf16 adjacency + folded weights (global_prefetch_b8)
    for (size_t off = (size_t)tid * 128; off < WS_BYTES; off += 256 * 128)
        __builtin_prefetch((const char*)wsB + off, 0, 1);

    // ---- phase 1: time -> sX (bf16), fre -> sT1 (bf16), latents -> sLat ----
    for (int i = tid; i < (NN * CC) / 4; i += 256) {
        int e = i << 2, r = e >> 7, c0 = e & 127;
        float4 v = ((const float4*)timeB)[i];
        Pack4 p; p.h[0] = f2bf(v.x); p.h[1] = f2bf(v.y); p.h[2] = f2bf(v.z); p.h[3] = f2bf(v.w);
        *(uint2*)&sX[r * LDX + c0] = p.q;
        v = ((const float4*)freB)[i];
        p.h[0] = f2bf(v.x); p.h[1] = f2bf(v.y); p.h[2] = f2bf(v.z); p.h[3] = f2bf(v.w);
        *(uint2*)&sT1[r * LDX + c0] = p.q;
    }
    for (int i = tid; i < 4 * CC; i += 256) sLat[i] = glat[i];
    __syncthreads();

    const float rsqc = 0.08838834764831845f;   // 128^-0.5

    // ---- phase 2a: logits[l][s] = (q_l . concat_s) / sqrt(C) ----
    for (int s = wave; s < 2 * NN; s += 8) {
        const unsigned short* row = (s < NN) ? (sX + s * LDX) : (sT1 + (s - NN) * LDX);
        float p0 = 0.f, p1 = 0.f, p2 = 0.f, p3 = 0.f;
        #pragma unroll
        for (int j = 0; j < 4; ++j) {
            int c = lane + 32 * j;
            float xv = bf2f(row[c]);
            p0 += xv * sLat[c];       p1 += xv * sLat[128 + c];
            p2 += xv * sLat[256 + c]; p3 += xv * sLat[384 + c];
        }
        p0 = wred32(p0); p1 = wred32(p1); p2 = wred32(p2); p3 = wred32(p3);
        if (lane == 0) {
            sLog[s] = p0 * rsqc;        sLog[512 + s]  = p1 * rsqc;
            sLog[1024 + s] = p2 * rsqc; sLog[1536 + s] = p3 * rsqc;
        }
    }
    __syncthreads();

    // ---- phase 2b: softmax over s (4 rows, one thread each) ----
    if (tid < 4) {
        float* rowp = sLog + tid * 512;
        float mx = -1e30f;
        for (int s = 0; s < 512; ++s) mx = fmaxf(mx, rowp[s]);
        float sum = 0.f;
        for (int s = 0; s < 512; ++s) { float e = __expf(rowp[s] - mx); rowp[s] = e; sum += e; }
        float inv = 1.0f / sum;
        for (int s = 0; s < 512; ++s) rowp[s] *= inv;
    }
    __syncthreads();

    // ---- phase 2c: fused[l][c] = sum_s p[l][s] * concat[s][c] ----
    {
        int c = tid & 127, l0 = tid >> 7;
        for (int l = l0; l < 4; l += 2) {
            float acc = 0.f;
            for (int s = 0; s < 512; ++s) {
                const unsigned short* row = (s < NN) ? (sX + s * LDX) : (sT1 + (s - NN) * LDX);
                acc += sLog[l * 512 + s] * bf2f(row[c]);
            }
            sFused[l * 128 + c] = acc;
        }
    }
    __syncthreads();

    // =========================== per-branch ===========================
    for (int br = 0; br < 2; ++br) {
        const float scale          = (br == 0) ? gscale_t[0] : gscale_f[0];
        const unsigned short* WT   = wsB + ((br == 0) ? WS_WT : WS_WF);
        const float* bias          = (br == 0) ? gbt : gbf;
        float* outB = gout + (size_t)br * BATCH * NN * CC + (size_t)b * NN * CC;

        if (br == 1) {  // reload fre into sX (sT1 was destroyed by branch 0)
            for (int i = tid; i < (NN * CC) / 4; i += 256) {
                int e = i << 2, r = e >> 7, c0 = e & 127;
                float4 v = ((const float4*)freB)[i];
                Pack4 p; p.h[0]=f2bf(v.x); p.h[1]=f2bf(v.y); p.h[2]=f2bf(v.z); p.h[3]=f2bf(v.w);
                *(uint2*)&sX[r * LDX + c0] = p.q;
            }
            __syncthreads();
        }

        // ---- phase 3: x += scale * softmax(x.fused^T/sqrt(C)) @ fused; build x^T ----
        for (int n = wave; n < NN; n += 8) {
            unsigned short* row = sX + n * LDX;
            float l0 = 0.f, l1 = 0.f, l2 = 0.f, l3 = 0.f;
            float xv[4];
            #pragma unroll
            for (int j = 0; j < 4; ++j) {
                int c = lane + 32 * j;
                float v = bf2f(row[c]); xv[j] = v;
                l0 += v * sFused[c];       l1 += v * sFused[128 + c];
                l2 += v * sFused[256 + c]; l3 += v * sFused[384 + c];
            }
            l0 = wred32(l0) * rsqc; l1 = wred32(l1) * rsqc;
            l2 = wred32(l2) * rsqc; l3 = wred32(l3) * rsqc;
            float mx = fmaxf(fmaxf(l0, l1), fmaxf(l2, l3));
            float e0 = __expf(l0 - mx), e1 = __expf(l1 - mx);
            float e2 = __expf(l2 - mx), e3 = __expf(l3 - mx);
            float inv = 1.0f / (e0 + e1 + e2 + e3);
            e0 *= inv; e1 *= inv; e2 *= inv; e3 *= inv;
            #pragma unroll
            for (int j = 0; j < 4; ++j) {
                int c = lane + 32 * j;
                float attn = e0 * sFused[c] + e1 * sFused[128 + c]
                           + e2 * sFused[256 + c] + e3 * sFused[384 + c];
                unsigned short h = f2bf(xv[j] + scale * attn);
                row[c] = h;
                sR2[c * LDT + n] = h;   // transposed copy
            }
        }
        __syncthreads();

        const int r  = lane & 15;
        const int hi = lane >> 4;
        const v8f vzero = {0.f,0.f,0.f,0.f,0.f,0.f,0.f,0.f};

        // ---- step A: t1 = adj @ x  (A fragment reused x4; acc block = 32 VGPRs) ----
        for (int pp = 0; pp < 2; ++pp) {
            const int i0 = (wave + 8 * pp) << 4;
            for (int jh = 0; jh < 2; ++jh) {
                v8f acc[4];
                #pragma unroll
                for (int j = 0; j < 4; ++j) acc[j] = vzero;
                for (int k0 = 0; k0 < NN; k0 += 32) {
                    FragBF a;
                    const unsigned short* ar = adjB + (i0 + r) * NN + k0 + (hi ? 8 : 0);
                    a.q[0] = *(const uint4*)(ar);
                    a.q[1] = *(const uint4*)(ar + 16);
                    #pragma unroll
                    for (int j = 0; j < 4; ++j) {
                        FragBF bm;
                        const unsigned short* brp =
                            sR2 + (((jh * 4 + j) << 4) + r) * LDT + k0 + (hi ? 8 : 0);
                        bm.q[0] = *(const uint4*)(brp);
                        bm.q[1] = *(const uint4*)(brp + 16);
                        acc[j] = __builtin_amdgcn_wmma_f32_16x16x32_bf16(
                                     false, a.v, false, bm.v, (short)0, acc[j], false, false);
                    }
                }
                const int m0 = i0 + hi * 8;
                #pragma unroll
                for (int j = 0; j < 4; ++j) {
                    const int j0 = (jh * 4 + j) << 4;
                    Pack8 pk;
                    #pragma unroll
                    for (int v = 0; v < 8; ++v) pk.h[v] = f2bf(acc[j][v]);
                    #pragma unroll
                    for (int v = 0; v < 8; ++v) sT1[(m0 + v) * LDX + j0 + r] = pk.h[v];
                    *(uint4*)&sT1T[(j0 + r) * LDT + m0] = pk.q;
                }
            }
        }
        __syncthreads();

        // ---- step B: u = adj @ t1  (B: t1^T from LDS) -> sR2 row-major ----
        for (int pp = 0; pp < 2; ++pp) {
            const int i0 = (wave + 8 * pp) << 4;
            for (int jh = 0; jh < 2; ++jh) {
                v8f acc[4];
                #pragma unroll
                for (int j = 0; j < 4; ++j) acc[j] = vzero;
                for (int k0 = 0; k0 < NN; k0 += 32) {
                    FragBF a;
                    const unsigned short* ar = adjB + (i0 + r) * NN + k0 + (hi ? 8 : 0);
                    a.q[0] = *(const uint4*)(ar);
                    a.q[1] = *(const uint4*)(ar + 16);
                    #pragma unroll
                    for (int j = 0; j < 4; ++j) {
                        FragBF bm;
                        const unsigned short* brp =
                            sT1T + (((jh * 4 + j) << 4) + r) * LDT + k0 + (hi ? 8 : 0);
                        bm.q[0] = *(const uint4*)(brp);
                        bm.q[1] = *(const uint4*)(brp + 16);
                        acc[j] = __builtin_amdgcn_wmma_f32_16x16x32_bf16(
                                     false, a.v, false, bm.v, (short)0, acc[j], false, false);
                    }
                }
                const int m0 = i0 + hi * 8;
                #pragma unroll
                for (int j = 0; j < 4; ++j) {
                    const int j0 = (jh * 4 + j) << 4;
                    #pragma unroll
                    for (int v = 0; v < 8; ++v)
                        sR2[(m0 + v) * LDX + j0 + r] = f2bf(acc[j][v]);
                }
            }
        }
        __syncthreads();

        // ---- step C: out = x + b + x@(W0-W2)^T' + t1@W1^T' + u@(2W2)^T' ----
        for (int pp = 0; pp < 2; ++pp) {
            const int i0 = (wave + 8 * pp) << 4;
            for (int jh = 0; jh < 2; ++jh) {
                v8f acc[4];
                #pragma unroll
                for (int j = 0; j < 4; ++j) acc[j] = vzero;
                for (int c0 = 0; c0 < CC; c0 += 32) {
                    const int aoff = c0 + (hi ? 8 : 0);
                    const int boff = c0 + (hi ? 8 : 0);
                    FragBF a0, a1, a2;
                    const unsigned short* ax = sX  + (i0 + r) * LDX + aoff;
                    const unsigned short* at = sT1 + (i0 + r) * LDX + aoff;
                    const unsigned short* au = sR2 + (i0 + r) * LDX + aoff;
                    a0.q[0] = *(const uint4*)(ax); a0.q[1] = *(const uint4*)(ax + 16);
                    a1.q[0] = *(const uint4*)(at); a1.q[1] = *(const uint4*)(at + 16);
                    a2.q[0] = *(const uint4*)(au); a2.q[1] = *(const uint4*)(au + 16);
                    #pragma unroll
                    for (int j = 0; j < 4; ++j) {
                        const int j0 = (jh * 4 + j) << 4;
                        FragBF b0, b1, b2;
                        const unsigned short* w0 = WT +         (j0 + r) * CC + boff;
                        const unsigned short* w1 = WT + 16384 + (j0 + r) * CC + boff;
                        const unsigned short* w2 = WT + 32768 + (j0 + r) * CC + boff;
                        b0.q[0] = *(const uint4*)(w0); b0.q[1] = *(const uint4*)(w0 + 16);
                        b1.q[0] = *(const uint4*)(w1); b1.q[1] = *(const uint4*)(w1 + 16);
                        b2.q[0] = *(const uint4*)(w2); b2.q[1] = *(const uint4*)(w2 + 16);
                        acc[j] = __builtin_amdgcn_wmma_f32_16x16x32_bf16(
                                     false, a0.v, false, b0.v, (short)0, acc[j], false, false);
                        acc[j] = __builtin_amdgcn_wmma_f32_16x16x32_bf16(
                                     false, a1.v, false, b1.v, (short)0, acc[j], false, false);
                        acc[j] = __builtin_amdgcn_wmma_f32_16x16x32_bf16(
                                     false, a2.v, false, b2.v, (short)0, acc[j], false, false);
                    }
                }
                const int m0 = i0 + hi * 8;
                #pragma unroll
                for (int j = 0; j < 4; ++j) {
                    const int j0 = (jh * 4 + j) << 4;
                    float bv = bias[j0 + r];
                    #pragma unroll
                    for (int v = 0; v < 8; ++v) {
                        float res = acc[j][v] + bf2f(sX[(m0 + v) * LDX + j0 + r]) + bv;
                        outB[(size_t)(m0 + v) * CC + j0 + r] = res;
                    }
                }
            }
        }
        __syncthreads();
    }
}

// ---------------------------------------------------------------------------
extern "C" void kernel_launch(void* const* d_in, const int* in_sizes, int n_in,
                              void* d_out, int out_size, void* d_ws, size_t ws_size,
                              hipStream_t stream) {
    const float* time_p = (const float*)d_in[0];
    const float* fre_p  = (const float*)d_in[1];
    const float* adj_p  = (const float*)d_in[2];
    const float* lat_p  = (const float*)d_in[3];
    const float* st_p   = (const float*)d_in[4];
    const float* sf_p   = (const float*)d_in[5];
    const float* Wt_p   = (const float*)d_in[6];
    const float* bt_p   = (const float*)d_in[7];
    const float* Wf_p   = (const float*)d_in[8];
    const float* bf_p   = (const float*)d_in[9];
    unsigned short* ws  = (unsigned short*)d_ws;
    float* out_p        = (float*)d_out;

    fl_prep<<<256, 256, 0, stream>>>(adj_p, Wt_p, Wf_p, ws);
    fl_main<<<BATCH, 256, 0, stream>>>(time_p, fre_p, lat_p, st_p, sf_p,
                                       bt_p, bf_p, ws, out_p);
}
